// SparseMoE_4767413699071
// MI455X (gfx1250) — compile-verified, matrix operand order checked by reading, not run on hardware
//
#include <hip/hip_runtime.h>
#include <hip/hip_bf16.h>
#include <math.h>

// Problem sizes (fixed by reference)
#define NTOK  16384   // B*T = 8*2048
#define DM    512     // model dim
#define NE    8       // experts
#define TOPK  2
#define KSTEPS (DM / 32)   // 16

typedef _Float16 v16h __attribute__((ext_vector_type(16)));
typedef float    v8f  __attribute__((ext_vector_type(8)));

#if __has_builtin(__builtin_amdgcn_sched_group_barrier)
#define SCHED_GROUP(mask, size, id) __builtin_amdgcn_sched_group_barrier((mask), (size), (id))
#else
#define SCHED_GROUP(mask, size, id)
#endif

__device__ __forceinline__ void atom_add_f32(float* p, float v) {
#if defined(__HIP_PLATFORM_AMD__)
    unsafeAtomicAdd(p, v);     // native global_atomic_add_f32
#else
    atomicAdd(p, v);
#endif
}

// ---------------------------------------------------------------------------
// Repack Wk [E][D][D] f32 -> f16, pre-swizzled into WMMA B-fragment order.
// Fragment linearization: idx = (((e*32 + ntile)*16 + kstep)*32 + lane),
// each idx owns 16 contiguous halfs (32B):
//   n  = ntile*16 + (lane&15)
//   kb = kstep*32 + (lane>>4)*8
//   halfs[0..7]  = W[kb+0 .. kb+7 ][n]
//   halfs[8..15] = W[kb+16.. kb+23][n]
// Matches the CDNA5 16-bit 32x16 B-matrix VGPR layout: a wave fetches a
// fragment with two coalesced global_load_b128 per lane.
// ---------------------------------------------------------------------------
__global__ __launch_bounds__(256) void convert_w_kernel(const float* __restrict__ Wk,
                                                        _Float16* __restrict__ whf) {
    int t = blockIdx.x * 256 + threadIdx.x;        // 131072 threads total
    int lane = t & 31;
    int ks   = (t >> 5)  & 15;
    int nt   = (t >> 9)  & 31;
    int e    =  t >> 14;
    int n    = nt * 16 + (lane & 15);
    int kb   = ks * 32 + ((lane >> 4) * 8);

    union { _Float16 h[16]; uint4 q[2]; } u;
    const float* W = Wk + (size_t)e * DM * DM;
#pragma unroll
    for (int j = 0; j < 8; ++j)
        u.h[j] = (_Float16)W[(size_t)(kb + j) * DM + n];
#pragma unroll
    for (int j = 0; j < 8; ++j)
        u.h[j + 8] = (_Float16)W[(size_t)(kb + 16 + j) * DM + n];

    uint4* dst = (uint4*)(whf + (size_t)t * 16);
    dst[0] = u.q[0];
    dst[1] = u.q[1];
}

// ---------------------------------------------------------------------------
// Router: one wave (32 lanes) per token.
//   logits = x@Wr + br ; noise = x@Wn + bn
//   noisy  = logits + eps * softplus(noise)
//   top-2 (first-index tie break, like lax.top_k), softmax over the two,
//   push (token, gate) into per-expert assignment lists.
// ---------------------------------------------------------------------------
__device__ __forceinline__ float softplus_stable(float z) {
    return fmaxf(z, 0.0f) + log1pf(expf(-fabsf(z)));
}

__global__ __launch_bounds__(256) void router_kernel(const float* __restrict__ x,
                                                     const float* __restrict__ eps,
                                                     const float* __restrict__ Wr,
                                                     const float* __restrict__ br,
                                                     const float* __restrict__ Wn,
                                                     const float* __restrict__ bn,
                                                     int*   __restrict__ counts,
                                                     int*   __restrict__ atok,
                                                     float* __restrict__ agate) {
    const int lane  = threadIdx.x & 31;
    const int wave  = threadIdx.x >> 5;
    const int token = blockIdx.x * 8 + wave;       // 2048 blocks * 8 waves
    if (token >= NTOK) return;

    const float* xr = x + (size_t)token * DM;
    float ar[NE], an[NE];
#pragma unroll
    for (int e = 0; e < NE; ++e) { ar[e] = 0.0f; an[e] = 0.0f; }

#pragma unroll 4
    for (int i = 0; i < DM / 32; ++i) {
        int d = lane + i * 32;
        float xv = xr[d];
        const float4* wr4 = (const float4*)(Wr + (size_t)d * NE);
        const float4* wn4 = (const float4*)(Wn + (size_t)d * NE);
        float4 r0 = wr4[0], r1 = wr4[1];
        float4 n0 = wn4[0], n1 = wn4[1];
        ar[0] += xv * r0.x; ar[1] += xv * r0.y; ar[2] += xv * r0.z; ar[3] += xv * r0.w;
        ar[4] += xv * r1.x; ar[5] += xv * r1.y; ar[6] += xv * r1.z; ar[7] += xv * r1.w;
        an[0] += xv * n0.x; an[1] += xv * n0.y; an[2] += xv * n0.z; an[3] += xv * n0.w;
        an[4] += xv * n1.x; an[5] += xv * n1.y; an[6] += xv * n1.z; an[7] += xv * n1.w;
    }

    // wave32 reduction
#pragma unroll
    for (int e = 0; e < NE; ++e) {
        float r = ar[e], n = an[e];
#pragma unroll
        for (int off = 16; off > 0; off >>= 1) {
            r += __shfl_xor(r, off, 32);
            n += __shfl_xor(n, off, 32);
        }
        ar[e] = r; an[e] = n;
    }

    if (lane == 0) {
        float noisy[NE];
#pragma unroll
        for (int e = 0; e < NE; ++e) {
            float logit = ar[e] + br[e];
            float nz    = an[e] + bn[e];
            noisy[e] = logit + eps[(size_t)token * NE + e] * softplus_stable(nz);
        }
        // top-2, first-index tie break (strict >)
        int i0 = 0; float v0 = noisy[0];
#pragma unroll
        for (int e = 1; e < NE; ++e)
            if (noisy[e] > v0) { v0 = noisy[e]; i0 = e; }
        int i1 = (i0 == 0) ? 1 : 0; float v1 = noisy[i1];
#pragma unroll
        for (int e = 0; e < NE; ++e)
            if (e != i0 && noisy[e] > v1) { v1 = noisy[e]; i1 = e; }

        // softmax over {v0, v1}, v0 >= v1
        float t  = expf(v1 - v0);
        float g0 = 1.0f / (1.0f + t);
        float g1 = t * g0;

        int p0 = atomicAdd(&counts[i0], 1);
        atok [i0 * NTOK + p0] = token;
        agate[i0 * NTOK + p0] = g0;
        int p1 = atomicAdd(&counts[i1], 1);
        atok [i1 * NTOK + p1] = token;
        agate[i1 * NTOK + p1] = g1;
    }
}

// ---------------------------------------------------------------------------
// Expert GEMM: grid = (1024 row-tiles, 8 experts), 128 threads = 4 waves.
// Block computes a 16-row (gathered tokens) x 512-col tile for expert e:
//   Y = cast_f16(Xg) @ Wh[e]   (f32 accumulate via v_wmma_f32_16x16x32_f16)
//   out[token] += gate * (Y + bk[e])   (native float atomics; exactly 2
//   commutative f32 adds per element onto zero -> bitwise deterministic)
// Wave w covers columns [w*128, w*128+128) = 8 N-tiles of 16.
//
// Software pipeline, enforced with sched_group_barrier so the scheduler
// cannot sink loads to just-before-use:
//   * B fragments: 4 rotating buffers, prefetch distance 3. At step j the
//     wave consumes b[j&3] while fragments j+1, j+2 are resident and j+3 is
//     in flight — all four mod-4 residues distinct, so RA must keep them.
//   * raw-A floats for ks+1 prefetched each iteration (index wrapped, no
//     branches -> single scheduling region).
//   * per k-step: 20 VMEM reads (16 B + 4 A) and 8 WMMAs, pinned to the
//     pattern 4x(3 loads, 1 WMMA) + 4x(2 loads, 1 WMMA).
// ---------------------------------------------------------------------------
__global__ __launch_bounds__(128) void moe_gemm_kernel(const float* __restrict__ x,
                                                       const _Float16* __restrict__ whf,
                                                       const int*   __restrict__ counts,
                                                       const int*   __restrict__ atok,
                                                       const float* __restrict__ agate,
                                                       const float* __restrict__ bk,
                                                       float* __restrict__ out) {
    const int e    = blockIdx.y;
    const int cnt  = counts[e];
    const int row0 = blockIdx.x * 16;
    if (row0 >= cnt) return;

    const int lane = threadIdx.x & 31;
    const int wave = threadIdx.x >> 5;
    const int hi   = lane >> 4;        // 0: K-lo half, 1: K-hi half
    const int nlo  = lane & 15;

    // A-fragment source row for this lane (M = lane&15), clamped for tail tiles
    const int rA   = row0 + nlo;
    const int idxA = e * NTOK + (rA < cnt ? rA : 0);
    const int tokA = atok[idxA];
    const float* ax = x + (size_t)tokA * DM + hi * 8;   // lane's K sub-chunk base

    // B fragment base for (e, ntile = wave*8, ks = 0), this lane.
    // Offset for (nt, ks): nt*8192 + ks*512 halfs.
    const _Float16* wbase = whf + (size_t)(e * 32 + wave * 8) * (16 * 32 * 16)
                                + (size_t)lane * 16;

    v8f acc[8] = {};  // 8 N-tiles x (16x16 f32 tile / 8 VGPRs)

    union BU { v16h v; uint4 q[2]; };
    BU b[4];

    // ---- pipeline prologue: fragments j = 0, 1, 2 (ks=0, nt=0..2) ----
    {
        const uint4* bp0 = (const uint4*)(wbase);
        b[0].q[0] = bp0[0]; b[0].q[1] = bp0[1];
        const uint4* bp1 = (const uint4*)(wbase + 8192);
        b[1].q[0] = bp1[0]; b[1].q[1] = bp1[1];
        const uint4* bp2 = (const uint4*)(wbase + 2 * 8192);
        b[2].q[0] = bp2[0]; b[2].q[1] = bp2[1];
    }
    float4 ra0 = ((const float4*)(ax))[0];
    float4 ra1 = ((const float4*)(ax))[1];
    float4 ra2 = ((const float4*)(ax + 16))[0];
    float4 ra3 = ((const float4*)(ax + 16))[1];

    for (int ks = 0; ks < KSTEPS; ++ks) {
        // Convert current raw A -> fragment:
        // halfs[0..7] = K kb..kb+7, halfs[8..15] = K kb+16..kb+23
        v16h a;
        a[0]  = (_Float16)ra0.x; a[1]  = (_Float16)ra0.y; a[2]  = (_Float16)ra0.z; a[3]  = (_Float16)ra0.w;
        a[4]  = (_Float16)ra1.x; a[5]  = (_Float16)ra1.y; a[6]  = (_Float16)ra1.z; a[7]  = (_Float16)ra1.w;
        a[8]  = (_Float16)ra2.x; a[9]  = (_Float16)ra2.y; a[10] = (_Float16)ra2.z; a[11] = (_Float16)ra2.w;
        a[12] = (_Float16)ra3.x; a[13] = (_Float16)ra3.y; a[14] = (_Float16)ra3.z; a[15] = (_Float16)ra3.w;

        // Prefetch next k-step's raw A (index wrapped: last iteration reloads
        // ks=0 harmlessly, keeping the body branch-free)
        {
            const int aks = (ks + 1) & (KSTEPS - 1);
            const float4* pa = (const float4*)(ax + aks * 32);
            ra0 = pa[0]; ra1 = pa[1];
            const float4* pb = (const float4*)(ax + aks * 32 + 16);
            ra2 = pb[0]; ra3 = pb[1];
        }

#pragma unroll
        for (int nt = 0; nt < 8; ++nt) {
            // Prefetch fragment j+3 (j = ks*8+nt) into b[(nt+3)&3]
            const int jnt = nt + 3;
            const int nks = (ks + (jnt >> 3)) & (KSTEPS - 1);   // wrap on tail
            const int nnt = jnt & 7;
            const uint4* bp = (const uint4*)(wbase + (size_t)nnt * 8192
                                                   + (size_t)nks * 512);
            b[(nt + 3) & 3].q[0] = bp[0];
            b[(nt + 3) & 3].q[1] = bp[1];

            acc[nt] = __builtin_amdgcn_wmma_f32_16x16x32_f16(
                false, a, false, b[nt & 3].v, (short)0, acc[nt], false, false);
        }

        // Pin the issue pattern: 20 VMEM reads interleaved with 8 WMMAs.
        // (literal args only: the builtin requires integer constant expressions)
        SCHED_GROUP(0x020, 3, 0); SCHED_GROUP(0x008, 1, 0);
        SCHED_GROUP(0x020, 3, 0); SCHED_GROUP(0x008, 1, 0);
        SCHED_GROUP(0x020, 3, 0); SCHED_GROUP(0x008, 1, 0);
        SCHED_GROUP(0x020, 3, 0); SCHED_GROUP(0x008, 1, 0);
        SCHED_GROUP(0x020, 2, 0); SCHED_GROUP(0x008, 1, 0);
        SCHED_GROUP(0x020, 2, 0); SCHED_GROUP(0x008, 1, 0);
        SCHED_GROUP(0x020, 2, 0); SCHED_GROUP(0x008, 1, 0);
        SCHED_GROUP(0x020, 2, 0); SCHED_GROUP(0x008, 1, 0);
    }

    // Epilogue: D layout — element r of acc maps to row (r + 8*hi), col (lane&15).
    // Branchless clamp: load unconditionally from a safe slot, select gate.
    float g[8]; int tk[8];
#pragma unroll
    for (int r = 0; r < 8; ++r) {
        int rr = row0 + r + 8 * hi;
        int rc = (rr < cnt) ? rr : 0;
        tk[r]    = atok [e * NTOK + rc];
        float gg = agate[e * NTOK + rc];
        g[r] = (rr < cnt) ? gg : 0.0f;
    }
#pragma unroll
    for (int nt = 0; nt < 8; ++nt) {
        int col = (wave * 8 + nt) * 16 + nlo;
        float bias = bk[e * DM + col];
#pragma unroll
        for (int r = 0; r < 8; ++r) {
            atom_add_f32(out + (size_t)tk[r] * DM + col, g[r] * (acc[nt][r] + bias));
        }
    }
}

// ---------------------------------------------------------------------------
// Launch
// ---------------------------------------------------------------------------
extern "C" void kernel_launch(void* const* d_in, const int* in_sizes, int n_in,
                              void* d_out, int out_size, void* d_ws, size_t ws_size,
                              hipStream_t stream) {
    const float* x   = (const float*)d_in[0];
    const float* eps = (const float*)d_in[1];
    const float* Wr  = (const float*)d_in[2];
    const float* br  = (const float*)d_in[3];
    const float* Wn  = (const float*)d_in[4];
    const float* bn  = (const float*)d_in[5];
    const float* Wk  = (const float*)d_in[6];
    const float* bk  = (const float*)d_in[7];
    float* out = (float*)d_out;

    // Workspace carve-up (~5.3 MB)
    uint8_t* ws = (uint8_t*)d_ws;
    size_t off = 0;
    _Float16* whf = (_Float16*)(ws + off); off += (size_t)NE * DM * DM * sizeof(_Float16); // 4 MB
    int* counts   = (int*)(ws + off);      off += 256;
    int* atok     = (int*)(ws + off);      off += (size_t)NE * NTOK * sizeof(int);         // 512 KB
    float* agate  = (float*)(ws + off);    off += (size_t)NE * NTOK * sizeof(float);       // 512 KB

    (void)hipMemsetAsync(out, 0, (size_t)NTOK * DM * sizeof(float), stream);
    (void)hipMemsetAsync(counts, 0, 256, stream);

    // One-shot weight repack: E*32*16*32 = 131072 threads
    convert_w_kernel<<<512, 256, 0, stream>>>(Wk, whf);

    // Router: one wave per token
    router_kernel<<<NTOK / 8, 256, 0, stream>>>(x, eps, Wr, br, Wn, bn,
                                                counts, atok, agate);

    // Expert GEMM: over-launched grid with data-dependent early exit
    moe_gemm_kernel<<<dim3(NTOK / 16, NE), 128, 0, stream>>>(
        x, whf, counts, atok, agate, bk, out);
}
